// RWKV_TimeMix_90048284328552
// MI455X (gfx1250) — compile-verified, matrix operand order checked by reading, not run on hardware
//
#include <hip/hip_runtime.h>
#include <hip/hip_bf16.h>

typedef __attribute__((ext_vector_type(16))) __bf16 v16bf;
typedef __attribute__((ext_vector_type(8)))  __bf16 v8bf;
typedef __attribute__((ext_vector_type(8)))  float  v8f;
typedef __attribute__((ext_vector_type(4)))  int    v4i;

#define B_  8
#define T_  2048
#define C_  1024
#define M_  (B_ * T_)         // 16384 rows
#define CHUNK_  128
#define NCHUNK_ (T_ / CHUNK_) // 16
#define LN_EPS_ 1e-5f

// ---- gfx1250 async global->LDS support (probe via __has_builtin) ----------
#if defined(__has_builtin)
#if __has_builtin(__builtin_amdgcn_global_load_async_to_lds_b128)
#define HAVE_ASYNC_LDS 1
#endif
#endif

#ifdef HAVE_ASYNC_LDS
typedef __attribute__((address_space(1))) v4i* gp_v4i;   // global int4*
typedef __attribute__((address_space(3))) v4i* lp_v4i;   // LDS int4*
__device__ __forceinline__ void async_copy_b128(const __bf16* g, __bf16* l) {
  __builtin_amdgcn_global_load_async_to_lds_b128((gp_v4i)g, (lp_v4i)l, 0, 0);
}
__device__ __forceinline__ void wait_async0() {
#if __has_builtin(__builtin_amdgcn_s_wait_asynccnt)
  __builtin_amdgcn_s_wait_asynccnt(0);
#else
  asm volatile("s_wait_asynccnt 0x0" ::: "memory");
#endif
}
#endif

// ---------------------------------------------------------------------------
// Elementwise: fp32 -> bf16 weight conversion
// ---------------------------------------------------------------------------
__global__ void cvt_f32_bf16(const float* __restrict__ src,
                             __bf16* __restrict__ dst, int n) {
  int i = blockIdx.x * blockDim.x + threadIdx.x;
  if (i < n) dst[i] = (__bf16)src[i];
}

// ---------------------------------------------------------------------------
// Token mix: xk/xv/xr = x*mix + x[t-1]*(1-mix), emitted in bf16 for WMMA
// ---------------------------------------------------------------------------
__global__ void mix_kernel(const float* __restrict__ x,
                           const float* __restrict__ tmk,
                           const float* __restrict__ tmv,
                           const float* __restrict__ tmr,
                           __bf16* __restrict__ xk,
                           __bf16* __restrict__ xv,
                           __bf16* __restrict__ xr) {
  int i = blockIdx.x * blockDim.x + threadIdx.x;   // exactly B*T*C threads
  int c = i & (C_ - 1);
  int t = (i >> 10) & (T_ - 1);
  float xc = x[i];
  float xp = (t > 0) ? x[i - C_] : 0.0f;
  float mk = tmk[c], mv = tmv[c], mr = tmr[c];
  xk[i] = (__bf16)(xc * mk + xp * (1.0f - mk));
  xv[i] = (__bf16)(xc * mv + xp * (1.0f - mv));
  xr[i] = (__bf16)(xc * mr + xp * (1.0f - mr));
}

// ---------------------------------------------------------------------------
// bf16 WMMA GEMM:  Out[m,n] = sum_k A[m,k] * W[n,k]   (A: MxK, W: NxK row-major)
// Block: 256 threads (8 waves), tile 256x128, K-step 32, double-buffered LDS.
// Wave grid 4(M) x 2(N); each wave owns 64x64 = 4x4 WMMA accumulators.
// Global->LDS via GLOBAL_LOAD_ASYNC_TO_LDS_B128 when available (ASYNCcnt).
// ---------------------------------------------------------------------------
#define BM_ 256
#define BN_ 128
#define BK_ 32

__global__ __launch_bounds__(256) void gemm_bf16_wmma(
    const __bf16* __restrict__ A, const __bf16* __restrict__ W,
    float* __restrict__ Out, int Kdim, int Ndim) {
  __shared__ __bf16 As[2][BM_][BK_];   // 2 x 16 KB
  __shared__ __bf16 Ws[2][BN_][BK_];   // 2 x  8 KB

  const int tid  = threadIdx.x;
  const int lane = tid & 31;
  const int wave = tid >> 5;
  const int wm   = wave >> 1;  // 0..3 -> M offset wm*64
  const int wn   = wave & 1;   // 0..1 -> N offset wn*64
  const size_t blockM = (size_t)blockIdx.x * BM_;
  const size_t blockN = (size_t)blockIdx.y * BN_;

  v8f acc[4][4];
#pragma unroll
  for (int mi = 0; mi < 4; ++mi)
#pragma unroll
    for (int ni = 0; ni < 4; ++ni)
      acc[mi][ni] = (v8f){0.f, 0.f, 0.f, 0.f, 0.f, 0.f, 0.f, 0.f};

  // tile-copy assignments: A = one 64B row per thread; B = 32B per thread
  const int arow = tid;              // 0..255
  const int brow = tid >> 1;         // 0..127
  const int bseg = (tid & 1) * 16;   // 0 or 16
  const __bf16* gA = A + (blockM + arow) * (size_t)Kdim;
  const __bf16* gW = W + (blockN + brow) * (size_t)Kdim + bseg;

  const int nk = Kdim / BK_;

  auto copy_tile = [&](int buf, int k0) {
#ifdef HAVE_ASYNC_LDS
#pragma unroll
    for (int q = 0; q < 4; ++q)
      async_copy_b128(gA + k0 + q * 8, &As[buf][arow][q * 8]);
#pragma unroll
    for (int q = 0; q < 2; ++q)
      async_copy_b128(gW + k0 + q * 8, &Ws[buf][brow][bseg + q * 8]);
#else
    v8bf a0 = *(const v8bf*)(gA + k0);
    v8bf a1 = *(const v8bf*)(gA + k0 + 8);
    v8bf a2 = *(const v8bf*)(gA + k0 + 16);
    v8bf a3 = *(const v8bf*)(gA + k0 + 24);
    v8bf w0 = *(const v8bf*)(gW + k0);
    v8bf w1 = *(const v8bf*)(gW + k0 + 8);
    *(v8bf*)&As[buf][arow][0]         = a0;
    *(v8bf*)&As[buf][arow][8]         = a1;
    *(v8bf*)&As[buf][arow][16]        = a2;
    *(v8bf*)&As[buf][arow][24]        = a3;
    *(v8bf*)&Ws[buf][brow][bseg]      = w0;
    *(v8bf*)&Ws[buf][brow][bseg + 8]  = w1;
#endif
  };

  copy_tile(0, 0);

  const int r  = lane & 15;          // fragment row within 16
  const int kb = (lane >> 4) * 8;    // per-lane K base (ISA 16-bit layout)

  for (int it = 0; it < nk; ++it) {
    const int cur = it & 1;
#ifdef HAVE_ASYNC_LDS
    wait_async0();                    // my copies into `cur` complete
#endif
    __syncthreads();                  // everyone copied `cur`; everyone done with cur^1

    if (it + 1 < nk) copy_tile(cur ^ 1, (it + 1) * BK_);

    v16bf afrag[4], bfrag[4];
#pragma unroll
    for (int mi = 0; mi < 4; ++mi) {
      const __bf16* p = &As[cur][wm * 64 + mi * 16 + r][kb];
      v8bf lo = *(const v8bf*)p;
      v8bf hi = *(const v8bf*)(p + 16);
      afrag[mi] = __builtin_shufflevector(lo, hi, 0, 1, 2, 3, 4, 5, 6, 7,
                                          8, 9, 10, 11, 12, 13, 14, 15);
    }
#pragma unroll
    for (int ni = 0; ni < 4; ++ni) {
      const __bf16* p = &Ws[cur][wn * 64 + ni * 16 + r][kb];
      v8bf lo = *(const v8bf*)p;
      v8bf hi = *(const v8bf*)(p + 16);
      bfrag[ni] = __builtin_shufflevector(lo, hi, 0, 1, 2, 3, 4, 5, 6, 7,
                                          8, 9, 10, 11, 12, 13, 14, 15);
    }

#pragma unroll
    for (int mi = 0; mi < 4; ++mi)
#pragma unroll
      for (int ni = 0; ni < 4; ++ni)
        acc[mi][ni] = __builtin_amdgcn_wmma_f32_16x16x32_bf16(
            false, afrag[mi], false, bfrag[ni], (short)0, acc[mi][ni],
            false, false);
  }

  // store D (fp32): VGPR j, lanes 0-15 -> M=j, lanes 16-31 -> M=j+8
  const int mhalf = (lane >> 4) * 8;
#pragma unroll
  for (int mi = 0; mi < 4; ++mi)
#pragma unroll
    for (int ni = 0; ni < 4; ++ni) {
      size_t col = blockN + wn * 64 + ni * 16 + r;
#pragma unroll
      for (int j = 0; j < 8; ++j) {
        size_t row = blockM + wm * 64 + mi * 16 + mhalf + j;
        Out[row * Ndim + col] = acc[mi][ni][j];
      }
    }
}

// ---------------------------------------------------------------------------
// WKV scan, pass A: per (b, chunk, c) local scan with zero init.
// gid layout: c fastest -> fully coalesced streaming at HBM roofline.
// Also materializes kv = k*v in-place over the v buffer.
// ---------------------------------------------------------------------------
__global__ void scan_chunk_partial(const float* __restrict__ kbuf,
                                   float* __restrict__ vbuf,      // becomes kv
                                   const float* __restrict__ td,  // [C] flat
                                   float* __restrict__ chunkP) {
  int gid = blockIdx.x * blockDim.x + threadIdx.x;  // B*NCHUNK*C
  int c  = gid & (C_ - 1);
  int jj = gid >> 10;              // b*NCHUNK + j
  int j  = jj & (NCHUNK_ - 1);
  int b  = jj >> 4;
  float decay = __expf(td[c]);
  size_t idx = ((size_t)b * T_ + (size_t)j * CHUNK_) * C_ + c;
  float s = 0.0f;
  for (int i = 0; i < CHUNK_; ++i, idx += C_) {
    float kvv = kbuf[idx] * vbuf[idx];
    vbuf[idx] = kvv;
    s = kvv + decay * s;
  }
  chunkP[gid] = s;
}

// Pass B: sequential combine of NCHUNK partials per channel (cheap).
__global__ void scan_combine(const float* __restrict__ chunkP,
                             const float* __restrict__ td,
                             float* __restrict__ chunkInit) {
  int ch = blockIdx.x * blockDim.x + threadIdx.x;  // B*C
  int c = ch & (C_ - 1);
  int b = ch >> 10;
  float dL = __expf(td[c] * (float)CHUNK_);        // decay^CHUNK
  float s = 0.0f;
  for (int j = 0; j < NCHUNK_; ++j) {
    int g = (((b << 4) + j) << 10) + c;
    chunkInit[g] = s;
    s = chunkP[g] + dL * s;
  }
}

// Pass C: re-scan each chunk from its true incoming state, emit o.
__global__ void scan_apply(const float* __restrict__ kvbuf,
                           const float* __restrict__ rbuf,
                           const float* __restrict__ td,
                           const float* __restrict__ tf,
                           const float* __restrict__ chunkInit,
                           float* __restrict__ obuf) {
  int gid = blockIdx.x * blockDim.x + threadIdx.x;  // B*NCHUNK*C
  int c  = gid & (C_ - 1);
  int jj = gid >> 10;
  int j  = jj & (NCHUNK_ - 1);
  int b  = jj >> 4;
  float decay = __expf(td[c]);
  float first = tf[c];
  size_t idx = ((size_t)b * T_ + (size_t)j * CHUNK_) * C_ + c;
  float s = chunkInit[gid];
  for (int i = 0; i < CHUNK_; ++i, idx += C_) {
    float kvv = kvbuf[idx];
    s = kvv + decay * s;
    obuf[idx] = rbuf[idx] * (first * kvv + s);
  }
}

// ---------------------------------------------------------------------------
// LayerNorm over C=1024 per row; emits bf16 for the final WMMA GEMM.
// ---------------------------------------------------------------------------
__global__ __launch_bounds__(256) void layernorm_bf16(
    const float* __restrict__ o, const float* __restrict__ g,
    const float* __restrict__ bb, __bf16* __restrict__ y) {
  __shared__ float r1[256];
  __shared__ float r2[256];
  int row = blockIdx.x;
  int tid = threadIdx.x;
  const float* p = o + (size_t)row * C_;
  float vals[4];
  float sum = 0.f, sq = 0.f;
#pragma unroll
  for (int e = 0; e < 4; ++e) {
    float v = p[tid + e * 256];
    vals[e] = v;
    sum += v;
    sq += v * v;
  }
  r1[tid] = sum;
  r2[tid] = sq;
  __syncthreads();
  for (int off = 128; off > 0; off >>= 1) {
    if (tid < off) { r1[tid] += r1[tid + off]; r2[tid] += r2[tid + off]; }
    __syncthreads();
  }
  float mean = r1[0] * (1.0f / C_);
  float var  = r2[0] * (1.0f / C_) - mean * mean;
  float rstd = rsqrtf(var + LN_EPS_);
#pragma unroll
  for (int e = 0; e < 4; ++e) {
    int c = tid + e * 256;
    y[(size_t)row * C_ + c] = (__bf16)((vals[e] - mean) * rstd * g[c] + bb[c]);
  }
}

// ---------------------------------------------------------------------------
// Launch
// ---------------------------------------------------------------------------
extern "C" void kernel_launch(void* const* d_in, const int* in_sizes, int n_in,
                              void* d_out, int out_size, void* d_ws, size_t ws_size,
                              hipStream_t stream) {
  const float* x   = (const float*)d_in[0];
  const float* tmk = (const float*)d_in[1];
  const float* tmv = (const float*)d_in[2];
  const float* tmr = (const float*)d_in[3];
  const float* td  = (const float*)d_in[4];   // [H,S] = C flat
  const float* tf  = (const float*)d_in[5];
  const float* Wk  = (const float*)d_in[6];
  const float* Wv  = (const float*)d_in[7];
  const float* Wr  = (const float*)d_in[8];
  const float* Wo  = (const float*)d_in[9];
  const float* lng = (const float*)d_in[10];
  const float* lnb = (const float*)d_in[11];
  float* out = (float*)d_out;

  char* ws = (char*)d_ws;
  const size_t MB = 1ull << 20;
  const int Wn = C_ * C_;           // 1M weight elements
  const int Xn = M_ * C_;           // 16.7M activation elements

  __bf16* wkb = (__bf16*)(ws + 0 * MB);
  __bf16* wvb = (__bf16*)(ws + 2 * MB);
  __bf16* wrb = (__bf16*)(ws + 4 * MB);
  __bf16* wob = (__bf16*)(ws + 6 * MB);
  __bf16* xk  = (__bf16*)(ws + 8 * MB);    // 32 MB
  __bf16* xv  = (__bf16*)(ws + 40 * MB);   // 32 MB
  __bf16* xr  = (__bf16*)(ws + 72 * MB);   // 32 MB
  float*  kf  = (float*)(ws + 104 * MB);   // 64 MB
  float*  vf  = (float*)(ws + 168 * MB);   // 64 MB (becomes kv)
  float*  rf  = (float*)(ws + 232 * MB);   // 64 MB
  float*  chunkP    = (float*)(ws + 296 * MB);             // 512 KB
  float*  chunkInit = (float*)(ws + 296 * MB + 512 * 1024);
  float*  obuf = (float*)(ws + 8 * MB);    // reuse xk+xv region (64 MB)
  __bf16* ybuf = (__bf16*)(ws + 72 * MB);  // reuse xr region (32 MB)

  // 1) weight conversion to bf16
  cvt_f32_bf16<<<(Wn + 255) / 256, 256, 0, stream>>>(Wk, wkb, Wn);
  cvt_f32_bf16<<<(Wn + 255) / 256, 256, 0, stream>>>(Wv, wvb, Wn);
  cvt_f32_bf16<<<(Wn + 255) / 256, 256, 0, stream>>>(Wr, wrb, Wn);
  cvt_f32_bf16<<<(Wn + 255) / 256, 256, 0, stream>>>(Wo, wob, Wn);

  // 2) token mix -> bf16 activations
  mix_kernel<<<Xn / 256, 256, 0, stream>>>(x, tmk, tmv, tmr, xk, xv, xr);

  // 3) three WMMA projections
  dim3 ggrid(M_ / BM_, C_ / BN_);   // 64 x 8
  gemm_bf16_wmma<<<ggrid, 256, 0, stream>>>(xk, wkb, kf, C_, C_);
  gemm_bf16_wmma<<<ggrid, 256, 0, stream>>>(xv, wvb, vf, C_, C_);
  gemm_bf16_wmma<<<ggrid, 256, 0, stream>>>(xr, wrb, rf, C_, C_);

  // 4) chunked WKV scan (A: local partials, B: combine, C: apply)
  const int nscan = B_ * C_ * NCHUNK_;  // 131072
  scan_chunk_partial<<<nscan / 256, 256, 0, stream>>>(kf, vf, td, chunkP);
  scan_combine<<<(B_ * C_) / 256, 256, 0, stream>>>(chunkP, td, chunkInit);
  scan_apply<<<nscan / 256, 256, 0, stream>>>(vf, rf, td, tf, chunkInit, obuf);

  // 5) layernorm -> bf16
  layernorm_bf16<<<M_, 256, 0, stream>>>(obuf, lng, lnb, ybuf);

  // 6) output projection
  gemm_bf16_wmma<<<ggrid, 256, 0, stream>>>(ybuf, wob, out, C_, C_);

  (void)in_sizes; (void)n_in; (void)out_size; (void)ws_size;
}
